// AttentionCacheAdapter_90769838833686
// MI455X (gfx1250) — matrix operand order repaired and checked by reading, not compiled, not run on hardware
//
#include <hip/hip_runtime.h>
#include <hip/hip_bf16.h>

typedef __attribute__((ext_vector_type(16))) _Float16 v16h;
typedef __attribute__((ext_vector_type(8)))  float    v8f;

// Problem constants (from reference)
constexpr int B_   = 8;
constexpr int H_   = 32;
constexpr int HKV_ = 8;
constexpr int HD_  = 128;
constexpr int S_   = 4096;
constexpr int D_   = H_ * HD_;     // 4096
constexpr int DKV_ = HKV_ * HD_;   // 1024
constexpr int G_   = H_ / HKV_;    // 4
constexpr int NPAIR = B_ * HKV_;   // 64
constexpr int NSPLIT = 16;                 // flash-decoding splits over S
constexpr int SPLIT_LEN = S_ / NSPLIT;     // 256 keys per split
constexpr int NTILE = SPLIT_LEN / 16;      // 16 key-tiles of 16 per split
constexpr float SCALE_ = 0.08838834764831845f;   // 1/sqrt(128)

// ---------------------------------------------------------------------------
// Kernel 1: fused QKV projection.  hidden (8 x 4096) @ [Wq|Wk|Wv].
// Weights (100 MB) are the traffic; each element is read exactly once.
// hidden is tiled through 32KB LDS and broadcast across the 8 batch rows.
// ---------------------------------------------------------------------------
__global__ void __launch_bounds__(256)
qkv_proj(const float* __restrict__ hid,
         const float* __restrict__ Wq, const float* __restrict__ Wk,
         const float* __restrict__ Wv,
         float* __restrict__ qw, float* __restrict__ kw, float* __restrict__ vw) {
  __shared__ float hs[B_ * 1024];
  const int col = blockIdx.x * 256 + threadIdx.x;  // 0..6143
  float acc[B_];
#pragma unroll
  for (int b = 0; b < B_; ++b) acc[b] = 0.f;

  const float* wbase;
  int stride, wcol;
  if (col < D_)              { wbase = Wq; stride = D_;   wcol = col; }
  else if (col < D_ + DKV_)  { wbase = Wk; stride = DKV_; wcol = col - D_; }
  else                       { wbase = Wv; stride = DKV_; wcol = col - D_ - DKV_; }

  for (int kt = 0; kt < 4; ++kt) {
    __syncthreads();
    for (int i = threadIdx.x; i < B_ * 1024; i += 256)
      hs[i] = hid[(i >> 10) * D_ + kt * 1024 + (i & 1023)];
    __syncthreads();
    const float* wp = wbase + (size_t)(kt * 1024) * stride + wcol;
    __builtin_prefetch(wp + (size_t)256 * stride, 0, 1);
    for (int k = 0; k < 1024; ++k) {
      const float w = wp[(size_t)k * stride];
#pragma unroll
      for (int b = 0; b < B_; ++b) acc[b] = fmaf(hs[b * 1024 + k], w, acc[b]);
    }
  }

  if (col < D_) {
#pragma unroll
    for (int b = 0; b < B_; ++b) qw[(size_t)b * D_ + col] = acc[b];
  } else if (col < D_ + DKV_) {
#pragma unroll
    for (int b = 0; b < B_; ++b) kw[(size_t)b * DKV_ + (col - D_)] = acc[b];
  } else {
#pragma unroll
    for (int b = 0; b < B_; ++b) vw[(size_t)b * DKV_ + (col - D_ - DKV_)] = acc[b];
  }
}

// ---------------------------------------------------------------------------
// Kernel 2: RoPE on q and k (in-place in workspace). One thread owns the
// (d, d+64) pair so the in-place update is race-free.
// ---------------------------------------------------------------------------
__global__ void __launch_bounds__(256)
rope(float* __restrict__ qw, float* __restrict__ kw,
     const float* __restrict__ cosp, const float* __restrict__ sinp) {
  const int idx = blockIdx.x * 256 + threadIdx.x;
  const int QP = B_ * H_ * 64;    // 16384 q pairs
  const int KP = B_ * HKV_ * 64;  //  4096 k pairs
  if (idx >= QP + KP) return;
  float* ptr;
  int b, d;
  size_t base;
  if (idx < QP) {
    b = idx / (H_ * 64);
    const int h = (idx / 64) % H_;
    d = idx & 63;
    base = (size_t)(b * H_ + h) * HD_;
    ptr = qw;
  } else {
    const int j = idx - QP;
    b = j / (HKV_ * 64);
    const int kvh = (j / 64) % HKV_;
    d = j & 63;
    base = (size_t)(b * HKV_ + kvh) * HD_;
    ptr = kw;
  }
  const float c0 = cosp[b * HD_ + d], c1 = cosp[b * HD_ + d + 64];
  const float s0 = sinp[b * HD_ + d], s1 = sinp[b * HD_ + d + 64];
  const float x1 = ptr[base + d], x2 = ptr[base + d + 64];
  ptr[base + d]      = x1 * c0 - x2 * s0;   // q*c + (-x2)*s
  ptr[base + d + 64] = x2 * c1 + x1 * s1;   // q*c + ( x1)*s
}

// ---------------------------------------------------------------------------
// Kernel 3: flash-decoding partial over the quantized KV cache.
// Grid: (NSPLIT, NPAIR), one wave32 per block (EXEC all-ones for WMMA).
// Scores: Q(16x128,f16, heads padded to 16) x K^T via v_wmma_f32_16x16x32_f16,
// 4 K-steps over HD=128 per 16-key tile. Online softmax + VALU P.V.
// ---------------------------------------------------------------------------
__global__ void __launch_bounds__(32)
flash_partial(const int* __restrict__ kq, const int* __restrict__ vq,
              const float* __restrict__ ksc, const float* __restrict__ kze,
              const float* __restrict__ vsc, const float* __restrict__ vze,
              const float* __restrict__ qw,
              float* __restrict__ part_ml, float* __restrict__ part_O) {
  const int split = blockIdx.x;
  const int pair  = blockIdx.y;             // b*HKV + kv
  const int b  = pair / HKV_;
  const int kv = pair % HKV_;
  const int lane = threadIdx.x;
  const int r  = lane & 15;
  const bool hi = lane >= 16;

  __shared__ float lds_s[4 * 16];
  __shared__ float lds_e[4 * 16];

  // --- Build A operand (Q tile, 16 rows x 128 K in 4 steps of 32), contiguous k order:
  //     lanes 0-15 : K 0..15;  lanes 16-31 : K 16..31.
  v16h a[4];
#pragma unroll
  for (int s = 0; s < 4; ++s)
#pragma unroll
    for (int j = 0; j < 16; ++j) a[s][j] = (_Float16)0.f;
  if (r < G_) {
    const float* qp = qw + (size_t)(b * H_ + kv * G_ + r) * HD_;
#pragma unroll
    for (int s = 0; s < 4; ++s) {
      const int b0 = 32 * s + (hi ? 16 : 0);
      const int b1 = 32 * s + 8 + (hi ? 16 : 0);
#pragma unroll
      for (int j = 0; j < 8; ++j) a[s][j]     = (_Float16)(qp[b0 + j] * SCALE_);
#pragma unroll
      for (int j = 0; j < 8; ++j) a[s][8 + j] = (_Float16)(qp[b1 + j] * SCALE_);
    }
  }

  float m[4], l[4], acc[4][4];
#pragma unroll
  for (int h = 0; h < 4; ++h) {
    m[h] = -1e30f;
    l[h] = 0.f;
#pragma unroll
    for (int j = 0; j < 4; ++j) acc[h][j] = 0.f;
  }

  const size_t pbase = (size_t)pair * S_;

  for (int t = 0; t < NTILE; ++t) {
    const int p0 = split * SPLIT_LEN + t * 16;

    // --- B operand: lane (mod 16) = key position; 16 contiguous f16 channels
    //     per lane (lanes 0-15: K 0..15, lanes 16-31: K 16..31 of each step).
    const int p = p0 + r;
    const float kzs = kze[pbase + p];
    const float kss = ksc[pbase + p];
    const int* kp = kq + (pbase + p) * HD_;

    v8f c = {};
#pragma unroll
    for (int s = 0; s < 4; ++s) {
      const int4* k4 = (const int4*)(kp + 32 * s + (hi ? 16 : 0));
      const int4 w0 = k4[0], w1 = k4[1], w2 = k4[2], w3 = k4[3];
      v16h bv;
#define CVK(x) ((_Float16)(((float)(x) - kzs) * kss))
      bv[0]  = CVK(w0.x); bv[1]  = CVK(w0.y); bv[2]  = CVK(w0.z); bv[3]  = CVK(w0.w);
      bv[4]  = CVK(w1.x); bv[5]  = CVK(w1.y); bv[6]  = CVK(w1.z); bv[7]  = CVK(w1.w);
      bv[8]  = CVK(w2.x); bv[9]  = CVK(w2.y); bv[10] = CVK(w2.z); bv[11] = CVK(w2.w);
      bv[12] = CVK(w3.x); bv[13] = CVK(w3.y); bv[14] = CVK(w3.z); bv[15] = CVK(w3.w);
#undef CVK
      c = __builtin_amdgcn_wmma_f32_16x16x32_f16(
              /*neg_a=*/false, a[s], /*neg_b=*/false, bv,
              /*c_mod=*/(short)0, c, /*reuse_a=*/false, /*reuse_b=*/false);
    }

    // D layout: lane(0-15)=key position, VGPR h = head h (heads 0..3 valid).
    __syncthreads();
    if (!hi) {
#pragma unroll
      for (int h = 0; h < 4; ++h) lds_s[h * 16 + r] = c[h];
    }
    __syncthreads();

    // Online softmax update (state replicated in every lane -> uniform).
    float nm[4], corr[4];
#pragma unroll
    for (int h = 0; h < 4; ++h) {
      float tm = -1e30f;
      for (int p2 = 0; p2 < 16; ++p2) tm = fmaxf(tm, lds_s[h * 16 + p2]);
      nm[h] = fmaxf(m[h], tm);
      corr[h] = __expf(m[h] - nm[h]);
      m[h] = nm[h];
    }
    if (!hi) {
#pragma unroll
      for (int h = 0; h < 4; ++h)
        lds_e[h * 16 + r] = __expf(lds_s[h * 16 + r] - nm[h]);
    }
    __syncthreads();

#pragma unroll
    for (int h = 0; h < 4; ++h) {
      float se = 0.f;
      for (int p2 = 0; p2 < 16; ++p2) se += lds_e[h * 16 + p2];
      l[h] = l[h] * corr[h] + se;
#pragma unroll
      for (int j = 0; j < 4; ++j) acc[h][j] *= corr[h];
    }

    // P.V : lane owns 4 output channels (4*lane .. 4*lane+3).
    for (int p2 = 0; p2 < 16; ++p2) {
      const size_t gp = pbase + p0 + p2;
      const float vs = vsc[gp], vz = vze[gp];
      const int4 vi = *(const int4*)(vq + gp * HD_ + 4 * lane);
      const float vf0 = ((float)vi.x - vz) * vs;
      const float vf1 = ((float)vi.y - vz) * vs;
      const float vf2 = ((float)vi.z - vz) * vs;
      const float vf3 = ((float)vi.w - vz) * vs;
#pragma unroll
      for (int h = 0; h < 4; ++h) {
        const float e = lds_e[h * 16 + p2];
        acc[h][0] = fmaf(e, vf0, acc[h][0]);
        acc[h][1] = fmaf(e, vf1, acc[h][1]);
        acc[h][2] = fmaf(e, vf2, acc[h][2]);
        acc[h][3] = fmaf(e, vf3, acc[h][3]);
      }
    }
  }

  // Partials: m,l per head + O[4][128].
  const size_t obase = (size_t)(pair * NSPLIT + split);
  if (lane < 4) {
    part_ml[obase * 8 + lane * 2 + 0] = m[lane];
    part_ml[obase * 8 + lane * 2 + 1] = l[lane];
  }
#pragma unroll
  for (int h = 0; h < 4; ++h)
#pragma unroll
    for (int j = 0; j < 4; ++j)
      part_O[(obase * 4 + h) * HD_ + 4 * lane + j] = acc[h][j];
}

// ---------------------------------------------------------------------------
// Kernel 4: merge split partials + new-token (RoPE'd k/v) contribution.
// One block per (b,h); thread = output channel d.
// ---------------------------------------------------------------------------
__global__ void __launch_bounds__(128)
combine(const float* __restrict__ qw, const float* __restrict__ kw,
        const float* __restrict__ vw,
        const float* __restrict__ part_ml, const float* __restrict__ part_O,
        float* __restrict__ attn) {
  const int blk = blockIdx.x;        // b*H + h
  const int b = blk >> 5, h = blk & 31;
  const int kv = h >> 2, hg = h & 3;
  const int d = threadIdx.x;
  __shared__ float red[128];

  // s_new = (q . k_new) * scale  via LDS tree reduction
  red[d] = qw[(size_t)(b * H_ + h) * HD_ + d] *
           kw[(size_t)(b * HKV_ + kv) * HD_ + d];
  __syncthreads();
  for (int off = 64; off > 0; off >>= 1) {
    if (d < off) red[d] += red[d + off];
    __syncthreads();
  }
  const float s_new = red[0] * SCALE_;

  const int pair = b * HKV_ + kv;
  float mg = s_new;
  for (int i = 0; i < NSPLIT; ++i)
    mg = fmaxf(mg, part_ml[(size_t)(pair * NSPLIT + i) * 8 + hg * 2]);

  float accd = 0.f, lg = 0.f;
  for (int i = 0; i < NSPLIT; ++i) {
    const size_t base = (size_t)(pair * NSPLIT + i);
    const float mi = part_ml[base * 8 + hg * 2 + 0];
    const float li = part_ml[base * 8 + hg * 2 + 1];
    const float w = __expf(mi - mg);
    lg = fmaf(w, li, lg);
    accd = fmaf(w, part_O[(base * 4 + hg) * HD_ + d], accd);
  }
  const float en = __expf(s_new - mg);
  lg += en;
  accd = fmaf(en, vw[(size_t)(b * HKV_ + kv) * HD_ + d], accd);
  attn[(size_t)b * D_ + h * HD_ + d] = accd / lg;
}

// ---------------------------------------------------------------------------
// Kernel 5: output projection, attn (8 x 4096) @ Wo (4096 x 4096).
// Same LDS-broadcast structure as qkv_proj: Wo read exactly once.
// ---------------------------------------------------------------------------
__global__ void __launch_bounds__(256)
out_proj(const float* __restrict__ attn, const float* __restrict__ Wo,
         float* __restrict__ out) {
  __shared__ float as[B_ * 1024];
  const int col = blockIdx.x * 256 + threadIdx.x;  // 0..4095
  float acc[B_];
#pragma unroll
  for (int b = 0; b < B_; ++b) acc[b] = 0.f;

  for (int kt = 0; kt < 4; ++kt) {
    __syncthreads();
    for (int i = threadIdx.x; i < B_ * 1024; i += 256)
      as[i] = attn[(i >> 10) * D_ + kt * 1024 + (i & 1023)];
    __syncthreads();
    const float* wp = Wo + (size_t)(kt * 1024) * D_ + col;
    __builtin_prefetch(wp + (size_t)256 * D_, 0, 1);
    for (int k = 0; k < 1024; ++k) {
      const float w = wp[(size_t)k * D_];
#pragma unroll
      for (int b = 0; b < B_; ++b) acc[b] = fmaf(as[b * 1024 + k], w, acc[b]);
    }
  }
#pragma unroll
  for (int b = 0; b < B_; ++b) out[(size_t)b * D_ + col] = acc[b];
}

// ---------------------------------------------------------------------------
extern "C" void kernel_launch(void* const* d_in, const int* in_sizes, int n_in,
                              void* d_out, int out_size, void* d_ws, size_t ws_size,
                              hipStream_t stream) {
  (void)in_sizes; (void)n_in; (void)out_size; (void)ws_size;
  const float* hid  = (const float*)d_in[0];
  const float* Wq   = (const float*)d_in[1];
  const float* Wk   = (const float*)d_in[2];
  const float* Wv   = (const float*)d_in[3];
  const float* Wo   = (const float*)d_in[4];
  const float* cosp = (const float*)d_in[5];
  const float* sinp = (const float*)d_in[6];
  const int*   kq   = (const int*)d_in[7];
  const int*   vq   = (const int*)d_in[8];
  const float* ksc  = (const float*)d_in[9];
  const float* kze  = (const float*)d_in[10];
  const float* vsc  = (const float*)d_in[11];
  const float* vze  = (const float*)d_in[12];

  float* ws   = (float*)d_ws;
  float* qw   = ws;                        // B*D        = 32768
  float* kw   = qw + (size_t)B_ * D_;      // B*DKV      =  8192
  float* vw   = kw + (size_t)B_ * DKV_;    // B*DKV      =  8192
  float* attn = vw + (size_t)B_ * DKV_;    // B*D        = 32768
  float* pml  = attn + (size_t)B_ * D_;    // NPAIR*NSPLIT*8     =   8192
  float* pO   = pml + (size_t)NPAIR * NSPLIT * 8;  // NPAIR*NSPLIT*4*128 = 524288

  qkv_proj<<<24, 256, 0, stream>>>(hid, Wq, Wk, Wv, qw, kw, vw);
  rope<<<80, 256, 0, stream>>>(qw, kw, cosp, sinp);
  dim3 g2(NSPLIT, NPAIR);
  flash_partial<<<g2, 32, 0, stream>>>(kq, vq, ksc, kze, vsc, vze, qw, pml, pO);
  combine<<<B_ * H_, 128, 0, stream>>>(qw, kw, vw, pml, pO, attn);
  out_proj<<<16, 256, 0, stream>>>(attn, Wo, (float*)d_out);
}
